// NodeModel_65077344469531
// MI455X (gfx1250) — matrix operand order repaired and checked by reading, not compiled, not run on hardware
//
#include <hip/hip_runtime.h>
#include <hip/hip_fp16.h>
#include <math.h>

typedef __attribute__((ext_vector_type(16))) _Float16 v16h;
typedef __attribute__((ext_vector_type(8)))  _Float16 v8h;
typedef __attribute__((ext_vector_type(8)))  float    v8f;

#define N_NODE_F   128
#define N_EDGE_F   128
#define N_GLOBAL_F 64
#define IN_DIM     320
#define HIDDEN     128
#define LDS_K      328   // padded K stride (halves); multiple of 8 -> 16B-aligned b128 loads

// pack two f32 -> f16x2 and store as one b32 to LDS
__device__ __forceinline__ void st_h2(_Float16* p, float x, float y) {
    union { _Float16 h[2]; unsigned u; } t;
    t.h[0] = (_Float16)x;
    t.h[1] = (_Float16)y;
    *(unsigned*)p = t.u;
}

// ---------------------------------------------------------------------------
// Kernel 0: zero workspace (sums + counts) — required every launch, ws is poisoned
// ---------------------------------------------------------------------------
__global__ void zero_ws_kernel(float* __restrict__ ws, int n) {
    int i = blockIdx.x * blockDim.x + threadIdx.x;
    if (i < n) ws[i] = 0.0f;
}

// ---------------------------------------------------------------------------
// Kernel 1: scatter-add edge features to destination nodes.
// One wave32 per edge, each lane handles 4 consecutive floats (32*4 = 128).
// 819 MB streaming read -> HBM bound (~35us floor); atomics land in the
// 192 MB L2 (51 MB sums buffer is resident).
// ---------------------------------------------------------------------------
__global__ void scatter_edges_kernel(const float* __restrict__ edge_feats,
                                     const int*   __restrict__ edge_index,
                                     float* __restrict__ sums,
                                     float* __restrict__ counts,
                                     int n_edges) {
    long tid  = (long)blockIdx.x * blockDim.x + threadIdx.x;
    int  e    = (int)(tid >> 5);
    int  lane = (int)(tid & 31);
    if (e >= n_edges) return;

    int dst = edge_index[(size_t)n_edges + e];   // row 1 of edge_index = destinations
    const float4 v = *(const float4*)(edge_feats + (size_t)e * N_EDGE_F + lane * 4);
    float* s = sums + (size_t)dst * N_EDGE_F + lane * 4;
    atomicAdd(s + 0, v.x);
    atomicAdd(s + 1, v.y);
    atomicAdd(s + 2, v.z);
    atomicAdd(s + 3, v.w);
    if (lane == 0) atomicAdd(counts + dst, 1.0f);
}

// ---------------------------------------------------------------------------
// Kernel 2: fused mean / concat / GEMM (f16 WMMA, f32 accum) / shifted softplus.
// 1 wave = 16x128 output tile; K=320 -> 10 k-steps x 8 n-tiles of
// v_wmma_f32_16x16x32_f16. Per k-step all 8 B fragments are preloaded so the
// in-order DS returns let WMMAs overlap the remaining loads.
// ---------------------------------------------------------------------------
__launch_bounds__(128)
__global__ void node_mlp_kernel(const float* __restrict__ node_feats,
                                const float* __restrict__ global_feats,
                                const float* __restrict__ W,
                                const int*   __restrict__ batch,
                                const float* __restrict__ sums,
                                const float* __restrict__ counts,
                                float* __restrict__ out,
                                int n_nodes) {
    extern __shared__ _Float16 lds[];
    _Float16* Wl = lds;                       // HIDDEN x LDS_K  (f16 weights)
    _Float16* Xl = lds + HIDDEN * LDS_K;      // 4 waves x 16 x LDS_K (f16 activations)

    const int tid  = threadIdx.x;
    const int wave = tid >> 5;
    const int lane = tid & 31;

    // ---- stage W[128,320] -> LDS f16 (float4 global reads, b32 packed stores) ----
    for (int i = tid * 4; i < HIDDEN * IN_DIM; i += 128 * 4) {
        int row = i / IN_DIM;
        int col = i - row * IN_DIM;           // multiple of 4, never crosses a row
        float4 w4 = *(const float4*)(W + i);
        _Float16* d = Wl + row * LDS_K + col;
        st_h2(d,     w4.x, w4.y);
        st_h2(d + 2, w4.z, w4.w);
    }
    __syncthreads();

    _Float16* Xw = Xl + wave * 16 * LDS_K;    // per-wave private staging slab

    const int n_tiles     = (n_nodes + 15) >> 4;
    const int wave_global = blockIdx.x * 4 + wave;
    const int wave_step   = gridDim.x * 4;

    const int mrow  = lane & 15;
    const int hi    = lane >> 4;
    const int ahalf = hi * 8;    // A: high half-wave holds K+8..15 / K+24..31
    const int bhalf = hi * 8;    // B: high half-wave holds K+8..15 / K+24..31
    const int c2    = lane * 2;  // packed-pair column base for staging

    for (int tile = wave_global; tile < n_tiles; tile += wave_step) {
        const int node0 = tile * 16;

        // ---- stage X tile (16 x 320) -> LDS f16: [node | mean_edge | global] ----
        for (int m = 0; m < 16; ++m) {
            int node = node0 + m;
            if (node >= n_nodes) node = n_nodes - 1;   // safe clamp (6250*16 == 100000)
            const float* nf = node_feats   + (size_t)node * N_NODE_F;
            const float* sf = sums         + (size_t)node * N_EDGE_F;
            const float* gf = global_feats + (size_t)batch[node] * N_GLOBAL_F;
            const float invc = 1.0f / fmaxf(counts[node], 1.0f);
            _Float16* xr = Xw + m * LDS_K;
            #pragma unroll
            for (int j = 0; j < 2; ++j) {              // node feats, cols 0..127
                int c = c2 + 64 * j;
                float2 v = *(const float2*)(nf + c);
                st_h2(xr + c, v.x, v.y);
            }
            #pragma unroll
            for (int j = 0; j < 2; ++j) {              // mean edge, cols 128..255
                int c = c2 + 64 * j;
                float2 v = *(const float2*)(sf + c);
                st_h2(xr + 128 + c, v.x * invc, v.y * invc);
            }
            {                                          // global feats, cols 256..319
                float2 v = *(const float2*)(gf + c2);
                st_h2(xr + 256 + c2, v.x, v.y);
            }
        }
        // wave-private LDS region: compiler inserts s_wait_dscnt for the RAW dep

        v8f acc[8];
        #pragma unroll
        for (int nt = 0; nt < 8; ++nt) acc[nt] = (v8f)0.0f;

        #pragma unroll
        for (int ks = 0; ks < 10; ++ks) {
            const int k0 = ks * 32;

            // A fragment: 2 x ds_load_b128 per lane (ISA 16-bit A 16x32 layout)
            const _Float16* ab = Xw + mrow * LDS_K + k0 + ahalf;
            v8h a_lo = *(const v8h*)(ab);        // elements 0..7  -> K = k0+{0..7|8..15}
            v8h a_hi = *(const v8h*)(ab + 16);   // elements 8..15 -> K = k0+16+{...}
            v16h a = __builtin_shufflevector(a_lo, a_hi,
                                             0, 1, 2, 3, 4, 5, 6, 7,
                                             8, 9, 10, 11, 12, 13, 14, 15);

            // Preload all 8 B fragments for this k-step (in-order DS returns ->
            // partial s_wait_dscnt lets WMMAs overlap outstanding loads)
            v16h b[8];
            #pragma unroll
            for (int nt = 0; nt < 8; ++nt) {
                // B: lanes 0-15 hold K 0..7,16..23, lanes 16-31 hold K 8..15,24..31
                const _Float16* bb = Wl + (nt * 16 + mrow) * LDS_K + k0 + bhalf;
                v8h b_lo = *(const v8h*)(bb);
                v8h b_hi = *(const v8h*)(bb + 16);
                b[nt] = __builtin_shufflevector(b_lo, b_hi,
                                                0, 1, 2, 3, 4, 5, 6, 7,
                                                8, 9, 10, 11, 12, 13, 14, 15);
            }
            #pragma unroll
            for (int nt = 0; nt < 8; ++nt) {
                acc[nt] = __builtin_amdgcn_wmma_f32_16x16x32_f16(
                    /*neg_a=*/false, a, /*neg_b=*/false, b[nt],
                    /*c_mod=*/(short)0, acc[nt],
                    /*reuse_a=*/false, /*reuse_b=*/false);
            }
        }

        // ---- epilogue: shifted softplus + store (C/D layout: VGPR r -> row r|r+8) ----
        // softplus(x) - ln2 = log(0.5 + 0.5*exp(x)) : 2 trans ops + 1 fma
        const float ln2 = 0.6931471805599453f;
        #pragma unroll
        for (int nt = 0; nt < 8; ++nt) {
            #pragma unroll
            for (int r = 0; r < 8; ++r) {
                const int m    = r + (hi << 3);
                const int node = node0 + m;
                if (node < n_nodes) {
                    float x  = acc[nt][r];
                    float sp = (x > 15.0f) ? (x - ln2)
                                           : __logf(fmaf(__expf(x), 0.5f, 0.5f));
                    out[(size_t)node * HIDDEN + nt * 16 + mrow] = sp;
                }
            }
        }
    }
}

// ---------------------------------------------------------------------------
extern "C" void kernel_launch(void* const* d_in, const int* in_sizes, int n_in,
                              void* d_out, int out_size, void* d_ws, size_t ws_size,
                              hipStream_t stream) {
    const float* node_feats   = (const float*)d_in[0];
    const float* edge_feats   = (const float*)d_in[1];
    const float* global_feats = (const float*)d_in[2];
    const float* W            = (const float*)d_in[3];
    const int*   edge_index   = (const int*)d_in[4];
    const int*   batch        = (const int*)d_in[5];
    float*       out          = (float*)d_out;

    const int n_nodes = in_sizes[0] / N_NODE_F;
    const int n_edges = in_sizes[1] / N_EDGE_F;

    float* sums   = (float*)d_ws;                       // [n_nodes, 128]
    float* counts = sums + (size_t)n_nodes * N_EDGE_F;  // [n_nodes]

    // 0) zero sums+counts (contiguous)
    const int nzero = n_nodes * (N_EDGE_F + 1);
    zero_ws_kernel<<<(nzero + 255) / 256, 256, 0, stream>>>(sums, nzero);

    // 1) scatter edges: one wave per edge
    const long nthreads = (long)n_edges * 32;
    scatter_edges_kernel<<<(int)((nthreads + 255) / 256), 256, 0, stream>>>(
        edge_feats, edge_index, sums, counts, n_edges);

    // 2) fused mean/concat/GEMM/softplus
    const int shmem = (HIDDEN * LDS_K + 4 * 16 * LDS_K) * (int)sizeof(_Float16); // ~126 KB
    node_mlp_kernel<<<512, 128, shmem, stream>>>(
        node_feats, global_feats, W, batch, sums, counts, out, n_nodes);
}
